// GrassmannLM_60438779789933
// MI455X (gfx1250) — compile-verified, matrix operand order checked by reading, not run on hardware
//
#include <hip/hip_runtime.h>
#include <hip/hip_bf16.h>

// ---------------- model constants ----------------
#define GV    50257
#define GS    1024
#define GD    640
#define GNL   16
#define GDF   2560
#define GR    24
#define GDG   256
#define GPLU  276
#define GPLUP 288          // padded K for the gp1 GEMM (multiple of 32)
#define GB    4
#define GL    1024
#define GBL   (GB * GL)    // 4096

typedef __attribute__((ext_vector_type(16))) _Float16 v16h;
typedef __attribute__((ext_vector_type(8)))  float    v8f;

__device__ __forceinline__ int   iminD(int a, int b) { return a < b ? a : b; }
__device__ __forceinline__ float gelu_exact(float x) {
    return 0.5f * x * (1.0f + erff(x * 0.70710678118654752440f));
}

__device__ __forceinline__ v8f wmma16(const v16h a, const v16h b, const v8f c) {
    // (neg_a, A, neg_b, B, c_mod, C, reuse_a, reuse_b)
    return __builtin_amdgcn_wmma_f32_16x16x32_f16(false, a, false, b,
                                                  (short)0, c, false, false);
}

// Interior fragment loader: lane's 16 f16 values are two contiguous 8-float
// runs -> 4x global_load_b128, then packed converts. No guards, no branches.
__device__ __forceinline__ v16h frag_full(const float* __restrict__ row,
                                          int k0, int khi) {
    const float4* p0 = reinterpret_cast<const float4*>(row + k0 + khi);
    const float4* p1 = reinterpret_cast<const float4*>(row + k0 + khi + 16);
    const float4 x0 = p0[0], x1 = p0[1], y0 = p1[0], y1 = p1[1];
    v16h f;
    f[0]  = (_Float16)x0.x; f[1]  = (_Float16)x0.y;
    f[2]  = (_Float16)x0.z; f[3]  = (_Float16)x0.w;
    f[4]  = (_Float16)x1.x; f[5]  = (_Float16)x1.y;
    f[6]  = (_Float16)x1.z; f[7]  = (_Float16)x1.w;
    f[8]  = (_Float16)y0.x; f[9]  = (_Float16)y0.y;
    f[10] = (_Float16)y0.z; f[11] = (_Float16)y0.w;
    f[12] = (_Float16)y1.x; f[13] = (_Float16)y1.y;
    f[14] = (_Float16)y1.z; f[15] = (_Float16)y1.w;
    return f;
}

// Edge fragment loader: UNCONDITIONAL loads from clamped offsets, zeroed by
// select afterwards -> v_cndmask, no EXEC save/restore branches.
__device__ __forceinline__ v16h frag_edge(const float* __restrict__ rowClamped,
                                          bool rowOK, int k0, int khi, int K) {
    v16h f;
#pragma unroll
    for (int j = 0; j < 8; ++j) {
        const int k  = k0 + khi + ((j < 4) ? (2 * j) : (16 + 2 * (j - 4)));
        const float v0 = rowClamped[iminD(k,     K - 1)];
        const float v1 = rowClamped[iminD(k + 1, K - 1)];
        f[2 * j]     = (_Float16)((rowOK && (k     < K)) ? v0 : 0.0f);
        f[2 * j + 1] = (_Float16)((rowOK && (k + 1 < K)) ? v1 : 0.0f);
    }
    return f;
}

// Epilogue for one 16x16 tile. C layout: VGPR v -> row = tM + v + (lane>=16?8:0),
// col = tN + (lane&15). Rows always in range (M % 64 == 0, exact grid).
template <int MODE>
__device__ __forceinline__ void store_tile(const v8f c, int tM, int tN, int lane,
                                           const float* __restrict__ bias,
                                           float* __restrict__ Out,
                                           int N, int delta) {
    const int col   = tN + (lane & 15);
    const int rbase = tM + ((lane >> 4) << 3);
    if (col < N) {
        const float bv = bias ? bias[col] : 0.0f;
#pragma unroll
        for (int v = 0; v < 8; ++v) {
            const int  row = rbase + v;
            const float val = c[v] + bv;
            const long oi = (long)row * N + col;
            if (MODE == 0) {
                Out[oi] = val;
            } else if (MODE == 1) {
                Out[oi] = gelu_exact(val);
            } else if (MODE == 2) {
                Out[oi] = 1.0f / (1.0f + expf(-val));
            } else if (MODE == 3) {
                Out[oi] += val;
            } else { // MODE == 4: causal masked accumulation
                const int l = row & (GL - 1);
                if (l >= delta) Out[oi] += val;
            }
        }
    }
}

// =====================================================================
// WMMA GEMM: Out(M,N) = A(M,K) * W(N,K)^T + bias.
// Wave tile 32x32 (2x2 wmma tiles), block = 8 waves (2M x 4N) = 64x128.
// Requires M % 64 == 0 (true for all calls: M = 4096).
// MODE: 0=store 1=gelu 2=sigmoid 3=accumulate 4=causal-masked accumulate
// =====================================================================
template <int MODE>
__global__ void glm_gemm_wmma(const float* __restrict__ A,
                              const float* __restrict__ W,
                              const float* __restrict__ bias,
                              float* __restrict__ Out,
                              int M, int N, int K, int delta) {
    const int lane = threadIdx.x & 31;
    const int wave = threadIdx.x >> 5;          // 0..7
    const int wm   = wave & 1;                  // 2 waves along M
    const int wn   = wave >> 1;                 // 4 waves along N
    const int tileM = blockIdx.y * 64  + wm * 32;
    const int tileN = blockIdx.x * 128 + wn * 32;

    const int l15 = lane & 15;
    const int khi = (lane >> 4) << 3;           // 0 or 8

    const int m0 = tileM + l15;                 // always < M
    const int m1 = m0 + 16;
    const int n0 = tileN + l15;
    const int n1 = n0 + 16;
    const bool nOK0 = (n0 < N), nOK1 = (n1 < N);

    const float* arow0 = A + (long)m0 * K;
    const float* arow1 = A + (long)m1 * K;
    const float* wrow0 = W + (long)iminD(n0, N - 1) * K;
    const float* wrow1 = W + (long)iminD(n1, N - 1) * K;

    v8f c00 = {}, c01 = {}, c10 = {}, c11 = {};

    // Wave-uniform predicate -> scalar branch, EXEC stays full.
    if (((K & 31) == 0) && (tileN + 32 <= N)) {
        for (int k0 = 0; k0 < K; k0 += 32) {
            const v16h a0 = frag_full(arow0, k0, khi);
            const v16h a1 = frag_full(arow1, k0, khi);
            const v16h b0 = frag_full(wrow0, k0, khi);
            const v16h b1 = frag_full(wrow1, k0, khi);
            c00 = wmma16(a0, b0, c00);
            c01 = wmma16(a0, b1, c01);
            c10 = wmma16(a1, b0, c10);
            c11 = wmma16(a1, b1, c11);
        }
    } else {
        for (int k0 = 0; k0 < K; k0 += 32) {
            const v16h a0 = frag_edge(arow0, true, k0, khi, K);
            const v16h a1 = frag_edge(arow1, true, k0, khi, K);
            const v16h b0 = frag_edge(wrow0, nOK0, k0, khi, K);
            const v16h b1 = frag_edge(wrow1, nOK1, k0, khi, K);
            c00 = wmma16(a0, b0, c00);
            c01 = wmma16(a0, b1, c01);
            c10 = wmma16(a1, b0, c10);
            c11 = wmma16(a1, b1, c11);
        }
    }

    store_tile<MODE>(c00, tileM,      tileN,      lane, bias, Out, N, delta);
    store_tile<MODE>(c01, tileM,      tileN + 16, lane, bias, Out, N, delta);
    store_tile<MODE>(c10, tileM + 16, tileN,      lane, bias, Out, N, delta);
    store_tile<MODE>(c11, tileM + 16, tileN + 16, lane, bias, Out, N, delta);
}

// =====================================================================
// LayerNorm: one 256-thread block per row of D elements
// =====================================================================
__global__ void glm_layernorm(const float* __restrict__ X,
                              const float* __restrict__ w,
                              const float* __restrict__ b,
                              float* __restrict__ Y, int Dn) {
    const int row = blockIdx.x;
    const float* xr = X + (long)row * Dn;
    __shared__ float s1[256], s2[256];
    float sum = 0.0f, sq = 0.0f;
    for (int i = threadIdx.x; i < Dn; i += blockDim.x) {
        const float v = xr[i];
        sum += v; sq += v * v;
    }
    s1[threadIdx.x] = sum; s2[threadIdx.x] = sq;
    __syncthreads();
    for (int s = 128; s > 0; s >>= 1) {
        if (threadIdx.x < s) {
            s1[threadIdx.x] += s1[threadIdx.x + s];
            s2[threadIdx.x] += s2[threadIdx.x + s];
        }
        __syncthreads();
    }
    const float mean = s1[0] / (float)Dn;
    const float var  = s2[0] / (float)Dn - mean * mean;
    const float rstd = rsqrtf(var + 1e-5f);
    float* yr = Y + (long)row * Dn;
    for (int i = threadIdx.x; i < Dn; i += blockDim.x)
        yr[i] = (xr[i] - mean) * rstd * w[i] + b[i];
}

// =====================================================================
// Embedding: x[row, d] = tok_emb[ids[row], d] + pos_emb[row % L, d]
// =====================================================================
__global__ void glm_embed(const int* __restrict__ ids,
                          const float* __restrict__ tok_emb,
                          const float* __restrict__ pos_emb,
                          float* __restrict__ X) {
    const long idx = (long)blockIdx.x * blockDim.x + threadIdx.x;
    if (idx >= (long)GBL * GD) return;
    const int row = (int)(idx / GD);
    const int col = (int)(idx % GD);
    const int l   = row & (GL - 1);
    const int tok = ids[row];
    X[idx] = tok_emb[(long)tok * GD + col] + pos_emb[(long)l * GD + col];
}

__global__ void glm_zero(float* __restrict__ P, long n) {
    const long idx = (long)blockIdx.x * blockDim.x + threadIdx.x;
    if (idx < n) P[idx] = 0.0f;
}

// Pad gp1_w (NL,DG,276) -> (NL,DG,288) with zeros so the gp1 GEMM can use the
// interior b128 path. Run once per launch (deterministic).
__global__ void glm_pad_gp1w(const float* __restrict__ src,
                             float* __restrict__ dst) {
    const long idx = (long)blockIdx.x * blockDim.x + threadIdx.x;
    if (idx >= (long)GNL * GDG * GPLUP) return;
    const int  k  = (int)(idx % GPLUP);
    const long nl = idx / GPLUP;
    dst[idx] = (k < GPLU) ? src[nl * GPLU + k] : 0.0f;
}

// =====================================================================
// Plucker coordinates for one delta, L2-normalized (EPS floor), written with
// stride GPLUP (pad lanes zeroed). One 128-thread block per (b,l) row.
// =====================================================================
__global__ void glm_plucker(const float* __restrict__ Z,
                            float* __restrict__ P, int delta) {
    const int row = blockIdx.x;        // 0..GBL-1
    const int l   = row & (GL - 1);
    __shared__ float zc[GR], zp[GR], psh[GPLU], red[128];
    if (threadIdx.x < GR) {
        zc[threadIdx.x] = Z[(long)row * GR + threadIdx.x];
        zp[threadIdx.x] = (l >= delta) ? Z[(long)(row - delta) * GR + threadIdx.x]
                                       : 0.0f;
    }
    __syncthreads();
    float sq = 0.0f;
    for (int t = threadIdx.x; t < GPLU; t += 128) {
        int idx = t, i = 0, span = GR - 1;
        while (idx >= span) { idx -= span; --span; ++i; }
        const int j = i + 1 + idx;
        const float v = zp[i] * zc[j] - zp[j] * zc[i];
        psh[t] = v;
        sq += v * v;
    }
    red[threadIdx.x] = sq;
    __syncthreads();
    for (int s = 64; s > 0; s >>= 1) {
        if (threadIdx.x < s) red[threadIdx.x] += red[threadIdx.x + s];
        __syncthreads();
    }
    const float rn = 1.0f / fmaxf(sqrtf(red[0]), 1e-6f);
    for (int t = threadIdx.x; t < GPLU; t += 128)
        P[(long)row * GPLUP + t] = psh[t] * rn;
    if (threadIdx.x < GPLUP - GPLU)
        P[(long)row * GPLUP + GPLU + threadIdx.x] = 0.0f;
}

// =====================================================================
// g = geom / max(count,1); hg = concat(h, g) for the gate GEMM
// =====================================================================
__global__ void glm_build_g_hg(const float* __restrict__ geom,
                               const float* __restrict__ H,
                               float* __restrict__ G,
                               float* __restrict__ HG) {
    const long idx = (long)blockIdx.x * blockDim.x + threadIdx.x;
    if (idx >= (long)GBL * GD) return;
    const int row = (int)(idx / GD);
    const int col = (int)(idx % GD);
    const int l   = row & (GL - 1);
    float cnt = 0.0f;
#pragma unroll
    for (int s = 0; s < 7; ++s) cnt += (l >= (1 << s)) ? 1.0f : 0.0f;
    cnt = fmaxf(cnt, 1.0f);
    const float g = geom[idx] / cnt;
    G[idx] = g;
    HG[(long)row * (2 * GD) + col]      = H[idx];
    HG[(long)row * (2 * GD) + GD + col] = g;
}

// x += alpha*h + (1-alpha)*g
__global__ void glm_mix(float* __restrict__ X,
                        const float* __restrict__ ALPHA,
                        const float* __restrict__ H,
                        const float* __restrict__ G) {
    const long idx = (long)blockIdx.x * blockDim.x + threadIdx.x;
    if (idx >= (long)GBL * GD) return;
    const float a = ALPHA[idx];
    X[idx] += a * H[idx] + (1.0f - a) * G[idx];
}

// =====================================================================
// Host launcher
// =====================================================================
extern "C" void kernel_launch(void* const* d_in, const int* in_sizes, int n_in,
                              void* d_out, int out_size, void* d_ws, size_t ws_size,
                              hipStream_t stream) {
    (void)in_sizes; (void)n_in; (void)out_size; (void)ws_size;

    const int*   ids     = (const int*)  d_in[0];
    const float* tok_emb = (const float*)d_in[1];
    const float* pos_emb = (const float*)d_in[2];
    const float* ln1_w   = (const float*)d_in[3];
    const float* ln1_b   = (const float*)d_in[4];
    const float* red_w   = (const float*)d_in[5];
    const float* red_b   = (const float*)d_in[6];
    const float* gp1_w   = (const float*)d_in[7];
    const float* gp1_b   = (const float*)d_in[8];
    const float* gp2_w   = (const float*)d_in[9];
    const float* gp2_b   = (const float*)d_in[10];
    const float* gate_w  = (const float*)d_in[11];
    const float* gate_b  = (const float*)d_in[12];
    const float* ln2_w   = (const float*)d_in[13];
    const float* ln2_b   = (const float*)d_in[14];
    const float* fc1_w   = (const float*)d_in[15];
    const float* fc1_b   = (const float*)d_in[16];
    const float* fc2_w   = (const float*)d_in[17];
    const float* fc2_b   = (const float*)d_in[18];
    const float* fn_w    = (const float*)d_in[19];
    const float* fn_b    = (const float*)d_in[20];

    // ---- workspace carve-out (fp32) ----
    float* ws = (float*)d_ws;
    size_t o = 0;
    auto carve = [&](size_t nfl) { float* p = ws + o; o += nfl; return p; };
    float* x      = carve((size_t)GBL * GD);
    float* h      = carve((size_t)GBL * GD);
    float* z      = carve((size_t)GBL * GR);
    float* p      = carve((size_t)GBL * GPLUP);
    float* g1     = carve((size_t)GBL * GDG);
    float* geom   = carve((size_t)GBL * GD);
    float* g      = carve((size_t)GBL * GD);
    float* hg     = carve((size_t)GBL * 2 * GD);
    float* alpha  = carve((size_t)GBL * GD);
    float* h2     = carve((size_t)GBL * GD);
    float* ff     = carve((size_t)GBL * GDF);
    float* xf     = carve((size_t)GBL * GD);
    float* g1wpad = carve((size_t)GNL * GDG * GPLUP);

    const long totD  = (long)GBL * GD;
    const int  eBlks = (int)((totD + 255) / 256);
    const dim3 blk(256);
    auto gemmGrid = [](int M, int N) { return dim3((N + 127) / 128, M / 64); };

    // one-time-per-launch weight padding + embedding
    {
        const long n = (long)GNL * GDG * GPLUP;
        glm_pad_gp1w<<<(int)((n + 255) / 256), blk, 0, stream>>>(gp1_w, g1wpad);
    }
    glm_embed<<<eBlks, blk, 0, stream>>>(ids, tok_emb, pos_emb, x);

    for (int l = 0; l < GNL; ++l) {
        const float* rw  = red_w  + (long)l * GR  * GD;
        const float* rb  = red_b  + (long)l * GR;
        const float* g1w = g1wpad + (long)l * GDG * GPLUP;
        const float* g1b = gp1_b  + (long)l * GDG;
        const float* g2w = gp2_w  + (long)l * GD  * GDG;
        const float* g2b = gp2_b  + (long)l * GD;
        const float* gw  = gate_w + (long)l * GD * 2 * GD;
        const float* gb  = gate_b + (long)l * GD;
        const float* f1w = fc1_w  + (long)l * GDF * GD;
        const float* f1b = fc1_b  + (long)l * GDF;
        const float* f2w = fc2_w  + (long)l * GD * GDF;
        const float* f2b = fc2_b  + (long)l * GD;

        // --- Grassmann mixer ---
        glm_layernorm<<<GBL, blk, 0, stream>>>(x, ln1_w + (long)l * GD,
                                               ln1_b + (long)l * GD, h, GD);
        // z = h @ red_w^T + red_b : M=BL, N=24, K=640 (edge path, tiny)
        glm_gemm_wmma<0><<<gemmGrid(GBL, GR), blk, 0, stream>>>(
            h, rw, rb, z, GBL, GR, GD, 0);

        glm_zero<<<eBlks, blk, 0, stream>>>(geom, totD);
        for (int s = 0; s < 7; ++s) {
            const int delta = 1 << s;
            glm_plucker<<<GBL, dim3(128), 0, stream>>>(z, p, delta);
            // g1 = gelu(p @ gp1_w^T + b) : K=288 padded -> interior path
            glm_gemm_wmma<1><<<gemmGrid(GBL, GDG), blk, 0, stream>>>(
                p, g1w, g1b, g1, GBL, GDG, GPLUP, 0);
            // geom += mask(l>=delta) * (g1 @ gp2_w^T + b)
            glm_gemm_wmma<4><<<gemmGrid(GBL, GD), blk, 0, stream>>>(
                g1, g2w, g2b, geom, GBL, GD, GDG, delta);
        }
        glm_build_g_hg<<<eBlks, blk, 0, stream>>>(geom, h, g, hg);
        // alpha = sigmoid(hg @ gate_w^T + b) : K=1280
        glm_gemm_wmma<2><<<gemmGrid(GBL, GD), blk, 0, stream>>>(
            hg, gw, gb, alpha, GBL, GD, 2 * GD, 0);
        glm_mix<<<eBlks, blk, 0, stream>>>(x, alpha, h, g);

        // --- FeedForward ---
        glm_layernorm<<<GBL, blk, 0, stream>>>(x, ln2_w + (long)l * GD,
                                               ln2_b + (long)l * GD, h2, GD);
        glm_gemm_wmma<1><<<gemmGrid(GBL, GDF), blk, 0, stream>>>(
            h2, f1w, f1b, ff, GBL, GDF, GD, 0);
        glm_gemm_wmma<3><<<gemmGrid(GBL, GD), blk, 0, stream>>>(
            ff, f2w, f2b, x, GBL, GD, GDF, 0);
    }

    // final LN + tied LM head
    glm_layernorm<<<GBL, blk, 0, stream>>>(x, fn_w, fn_b, xf, GD);
    glm_gemm_wmma<0><<<gemmGrid(GBL, GV), blk, 0, stream>>>(
        xf, tok_emb, nullptr, (float*)d_out, GBL, GV, GD, 0);
}
